// Encoder_53025666236761
// MI455X (gfx1250) — compile-verified, matrix operand order checked by reading, not run on hardware
//
#include <hip/hip_runtime.h>
#include <hip/hip_bf16.h>
#include <math.h>

// ---------------- problem constants ----------------
#define SS 512   // sequence length
#define BB 512   // batch
#define DD 300   // input dim
#define HH 256   // hidden dim
#define LL 3     // layers

// fragment geometry (wave32 WMMA bf16 16x16x32)
#define FRAG_ELEMS 512            // 32 lanes x 16 bf16 per fragment
#define KT_H 8                    // 256 / 32 k-tiles for H-sized K
#define KT_X 10                   // ceil(300/32) k-tiles for D-sized K (zero padded)
#define NT   16                   // 256 / 16 n-tiles

// packed weight layout per direction (bf16 element offsets)
#define IH0_SZ (KT_X * NT * FRAG_ELEMS)          // 81920
#define HH_SZ  (KT_H * NT * FRAG_ELEMS)          // 65536
#define OFF_IH0 0
#define OFF_HH  (IH0_SZ)                         // 3 matrices
#define OFF_IH  (IH0_SZ + 3 * HH_SZ)             // 2 matrices
#define DIR_SZ  (IH0_SZ + 3 * HH_SZ + 2 * HH_SZ) // 409600 bf16 per direction

typedef __attribute__((ext_vector_type(16))) __bf16 v16bf;
typedef __attribute__((ext_vector_type(8)))  float  v8f;

// explicit global-address-space pointers -> global_load_b128 with saddr + 32-bit voffset
typedef __attribute__((address_space(1))) const __bf16* gbf_t;
typedef __attribute__((address_space(1))) const v16bf*  gv16_t;

__device__ __forceinline__ float fast_tanh(float v) {
#if __has_builtin(__builtin_amdgcn_tanhf)
    return __builtin_amdgcn_tanhf(v);   // gfx1250 v_tanh_f32 (TRANS, co-executes with WMMA)
#else
    return tanhf(v);
#endif
}

// K-position within a 32-wide k-tile for per-lane value index v (0..15) and lane half.
// Matches ISA 16-bit A-matrix 16x32 layout (CDNA5 ISA): lanes 0-15 half=0, 16-31 half=1.
__device__ __forceinline__ int kpos_of(int v, int half) {
    return (v < 8) ? (8 * half + v) : (16 + 8 * half + (v - 8));
}

// Inverse: for k-in-tile kk (0..31) -> (lane half, value index v)
__device__ __forceinline__ void inv_kpos(int kk, int& half, int& v) {
    half = (kk >> 3) & 1;
    v = (kk & 7) + ((kk & 16) >> 1);
}

// ---------------- kernel 1: pack f32 weights -> bf16 WMMA B-fragments ----------------
__global__ __launch_bounds__(256) void pack_weights(
    const float* __restrict__ fW_ih0, const float* __restrict__ fW_ih,
    const float* __restrict__ fW_hh,  const float* __restrict__ bW_ih0,
    const float* __restrict__ bW_ih,  const float* __restrict__ bW_hh,
    __bf16* __restrict__ wp)
{
    int e = blockIdx.x * blockDim.x + threadIdx.x;
    if (e >= 2 * DIR_SZ) return;
    int dir = e / DIR_SZ;
    int r   = e - dir * DIR_SZ;

    const float* Wih0 = dir ? bW_ih0 : fW_ih0;
    const float* Wih  = dir ? bW_ih  : fW_ih;
    const float* Whh  = dir ? bW_hh  : fW_hh;

    const float* src;
    int rr, kmax;
    if (r < IH0_SZ) {                       // layer-0 input weights (300 x 256), K padded to 320
        src = Wih0; rr = r; kmax = DD;
    } else if (r < IH0_SZ + 3 * HH_SZ) {    // recurrent weights, 3 layers (256 x 256)
        int q = r - IH0_SZ; int layer = q / HH_SZ;
        src = Whh + layer * HH * HH; rr = q - layer * HH_SZ; kmax = HH;
    } else {                                // inter-layer input weights, layers 1..2
        int q = r - (IH0_SZ + 3 * HH_SZ); int layer = q / HH_SZ;
        src = Wih + layer * HH * HH; rr = q - layer * HH_SZ; kmax = HH;
    }

    int fi = rr >> 9;                 // fragment index = kt*NT + nt
    int within = rr & 511;
    int lane = within >> 4, v = within & 15;
    int half = lane >> 4, nl = lane & 15;
    int kk = kpos_of(v, half);
    int kt = fi >> 4, nt = fi & 15;
    int k = kt * 32 + kk;
    int n = nt * 16 + nl;
    float val = (k < kmax) ? src[k * HH + n] : 0.0f;
    wp[e] = (__bf16)val;
}

// ---------------- GEMM micro-step, software-pipelined 1 deep ----------------
// C[16x16] x2 tiles += A(16 x 32*NKT) x W(32*NKT x 16) x2.
// wMat: uniform (SGPR) addrspace(1) base of the packed matrix.
// wOff: per-lane 32-bit element offset (nt-invariant part folded by caller = lane*16).
// All kt/nt displacements are compile-time constants -> folded into the 24-bit imm offset.
template <int NKT>
__device__ __forceinline__ void gemm_acc(const __bf16* __restrict__ ap,  // LDS, lane-adjusted
                                         gbf_t wMat, int wOff,
                                         int nt0, int nt1,
                                         v8f& c0, v8f& c1)
{
    v16bf a  = *(const v16bf*)(ap);
    v16bf b0 = *(gv16_t)(wMat + wOff + nt0 * FRAG_ELEMS);
    v16bf b1 = *(gv16_t)(wMat + wOff + nt1 * FRAG_ELEMS);
#pragma unroll
    for (int kt = 0; kt < NKT; ++kt) {
        v16bf an = a, b0n = b0, b1n = b1;
        if (kt + 1 < NKT) {   // prefetch next k-tile's fragments before issuing WMMAs
            an  = *(const v16bf*)(ap + (kt + 1) * FRAG_ELEMS);
            b0n = *(gv16_t)(wMat + wOff + ((kt + 1) * NT + nt0) * FRAG_ELEMS);
            b1n = *(gv16_t)(wMat + wOff + ((kt + 1) * NT + nt1) * FRAG_ELEMS);
        }
        c0 = __builtin_amdgcn_wmma_f32_16x16x32_bf16(false, a, false, b0, (short)0, c0, false, false);
        c1 = __builtin_amdgcn_wmma_f32_16x16x32_bf16(false, a, false, b1, (short)0, c1, false, false);
        a = an; b0 = b0n; b1 = b1n;
    }
}

// store one C element (row m, hidden col k) into LDS A-fragment layout as bf16
__device__ __forceinline__ void store_h_elem(__bf16* __restrict__ h, int m, int k, float val) {
    int kt = k >> 5, kk = k & 31;
    int half, v; inv_kpos(kk, half, v);
    int la = (half << 4) | m;
    h[kt * FRAG_ELEMS + la * 16 + v] = (__bf16)val;
}

// ---------------- kernel 2: persistent bidirectional stacked RNN ----------------
// grid: 64 blocks (= 2 dirs x 32 row-tiles), 256 threads (= 8 waves; wave w owns n-tiles 2w, 2w+1)
__global__ __launch_bounds__(256) void birnn_kernel(
    const float* __restrict__ x,      // [S, B, D]
    const float* __restrict__ fb,     // [L, H]
    const float* __restrict__ bbias,  // [L, H]
    const __bf16* __restrict__ wpk,   // packed weights, 2 * DIR_SZ
    float* __restrict__ out)          // [B, 2H]
{
    const int dir     = blockIdx.x & 1;
    const int rowtile = blockIdx.x >> 1;
    const int b0      = rowtile * 16;
    const int tid     = threadIdx.x;
    const int wave    = tid >> 5;
    const int lane    = tid & 31;
    const int nt0     = wave * 2, nt1 = nt0 + 1;
    const int nl      = lane & 15;
    const int halfc   = lane >> 4;
    const int wOff    = lane * 16;    // per-lane element offset inside any fragment

    __shared__ __align__(32) __bf16 sm_x[KT_X * FRAG_ELEMS];      // x A-fragments (padded K=320)
    __shared__ __align__(32) __bf16 sm_h[LL][KT_H * FRAG_ELEMS];  // hidden state, 3 layers

    // h(t=-1) = 0
    for (int i = tid; i < LL * KT_H * FRAG_ELEMS; i += 256)
        ((__bf16*)sm_h)[i] = (__bf16)0.0f;

    const float* bias = dir ? bbias : fb;
    float bia[LL][2];
#pragma unroll
    for (int l = 0; l < LL; ++l) {
        bia[l][0] = bias[l * HH + nt0 * 16 + nl];
        bia[l][1] = bias[l * HH + nt1 * 16 + nl];
    }

    // precompute t-invariant x-staging decode (20 elements per thread)
    int  srcOff[20];
    bool srcOk[20];
#pragma unroll
    for (int j = 0; j < 20; ++j) {
        int e = tid + j * 256;
        int kt = e >> 9, rem = e & 511;
        int la = rem >> 4, v = rem & 15;
        int half = la >> 4, m = la & 15;
        int k = kt * 32 + kpos_of(v, half);
        srcOk[j]  = (k < DD);
        srcOff[j] = (b0 + m) * DD + k;
    }

    const unsigned long long wBaseDir =
        (unsigned long long)(uintptr_t)wpk + (unsigned long long)dir * DIR_SZ * 2ull;

    // lane-adjusted LDS A-fragment base pointers
    const __bf16* apx = sm_x + wOff;
    const __bf16* aph[LL] = { sm_h[0] + wOff, sm_h[1] + wOff, sm_h[2] + wOff };

    __syncthreads();

    for (int t = 0; t < SS; ++t) {
        const int tt = dir ? (SS - 1 - t) : t;

        // Re-materialize the uniform (SGPR) weight base each step: blocks LICM from hoisting
        // ~800 VGPRs of loop-invariant fragment loads (which previously spilled to scratch).
        // Weights stay in L2 (1.6 MB shared by all 64 WGs). The asm "s" output keeps the base
        // uniform -> backend selects global_load_b128 saddr+voffset+imm form (no VALU addr math).
        unsigned long long wdir = wBaseDir;
        asm volatile("" : "+s"(wdir));
        gbf_t wg    = (gbf_t)(uintptr_t)wdir;
        gbf_t w_ih0 = wg + OFF_IH0;
        gbf_t w_hh  = wg + OFF_HH;
        gbf_t w_ih  = wg + OFF_IH;

        // stage x[tt] rows b0..b0+15 into LDS as bf16 A-fragments (zero-pad K >= 300)
        const float* xt = x + (size_t)tt * (BB * DD);
#pragma unroll
        for (int j = 0; j < 20; ++j) {
            float val = srcOk[j] ? xt[srcOff[j]] : 0.0f;
            sm_x[tid + j * 256] = (__bf16)val;
        }

        // prefetch next step's x row-block into caches (global_prefetch_b8)
        if (t + 1 < SS) {
            const int tn = dir ? (tt - 1) : (tt + 1);
            const char* pf = (const char*)(x + ((size_t)tn * BB + b0) * DD);
            if (tid * 128 < 16 * DD * 4)
                __builtin_prefetch(pf + tid * 128, 0, 3);
        }
        __syncthreads();

#pragma unroll
        for (int l = 0; l < LL; ++l) {
            v8f c0, c1;
#pragma unroll
            for (int r = 0; r < 8; ++r) { c0[r] = bia[l][0]; c1[r] = bia[l][1]; }

            // input-projection GEMM
            if (l == 0)
                gemm_acc<KT_X>(apx, w_ih0, wOff, nt0, nt1, c0, c1);
            else
                gemm_acc<KT_H>(aph[l - 1], w_ih + (l - 1) * HH_SZ, wOff, nt0, nt1, c0, c1);

            // recurrent GEMM (reads previous-step h[l])
            gemm_acc<KT_H>(aph[l], w_hh + l * HH_SZ, wOff, nt0, nt1, c0, c1);

            // activation: hardware v_tanh_f32
#pragma unroll
            for (int r = 0; r < 8; ++r) { c0[r] = fast_tanh(c0[r]); c1[r] = fast_tanh(c1[r]); }

            __syncthreads();  // all waves finished reading old h[l] / h[l-1]
#pragma unroll
            for (int r = 0; r < 8; ++r) {
                int m = r + 8 * halfc;
                store_h_elem(sm_h[l], m, nt0 * 16 + nl, c0[r]);
                store_h_elem(sm_h[l], m, nt1 * 16 + nl, c1[r]);
            }
            __syncthreads();

            // final output: layer 2 hidden at the last step, written in full f32 precision
            if (l == LL - 1 && t == SS - 1) {
#pragma unroll
                for (int r = 0; r < 8; ++r) {
                    int m = r + 8 * halfc;
                    out[(size_t)(b0 + m) * (2 * HH) + dir * HH + nt0 * 16 + nl] = c0[r];
                    out[(size_t)(b0 + m) * (2 * HH) + dir * HH + nt1 * 16 + nl] = c1[r];
                }
            }
        }
        __syncthreads();  // protect sm_x before next step restages it
    }
}

// ---------------- launch ----------------
extern "C" void kernel_launch(void* const* d_in, const int* in_sizes, int n_in,
                              void* d_out, int out_size, void* d_ws, size_t ws_size,
                              hipStream_t stream) {
    const float* x      = (const float*)d_in[0];
    const float* fW_ih0 = (const float*)d_in[1];
    const float* fW_ih  = (const float*)d_in[2];
    const float* fW_hh  = (const float*)d_in[3];
    const float* fb     = (const float*)d_in[4];
    const float* bW_ih0 = (const float*)d_in[5];
    const float* bW_ih  = (const float*)d_in[6];
    const float* bW_hh  = (const float*)d_in[7];
    const float* bb     = (const float*)d_in[8];

    __bf16* wp = (__bf16*)d_ws;  // 2 * DIR_SZ * 2 bytes = 1.6 MB of scratch

    const int pack_total = 2 * DIR_SZ;
    pack_weights<<<(pack_total + 255) / 256, 256, 0, stream>>>(
        fW_ih0, fW_ih, fW_hh, bW_ih0, bW_ih, bW_hh, wp);

    birnn_kernel<<<2 * (BB / 16), 256, 0, stream>>>(x, fb, bb, wp, (float*)d_out);
}
